// MainNet_33311766348149
// MI455X (gfx1250) — compile-verified, matrix operand order checked by reading, not run on hardware
//
#include <hip/hip_runtime.h>
#include <hip/hip_bf16.h>

// ---------- problem constants ----------
#define NB   128      // batch
#define NC   512      // channels
#define NHW  196      // H*W
#define NM   25088    // NB*NHW rows
#define NP   512      // codebook size

// ---------- workspace byte offsets ----------
#define OFF_WBF   0u            // 512*512 bf16     = 524288
#define OFF_W2    524288u       // 512 f32          =   2048
#define OFF_ASUM  526336u       // 128*512 f32      = 262144
#define OFF_AT    788480u       // 512*25088 bf16   = 25690112  (assign^T [P][M])
#define OFF_AMEAN 26478592u     // 512 f32          =   2048
#define OFF_XBFT  26480640u     // 512*25088 bf16   = 25690112  (x^T [C][M])

// ---------- output float offsets ----------
#define OUT_EMB   0         // [128,1024]
#define OUT_WSUM  131072    // [512,512]
#define OUT_WNUM  393216    // [512]

typedef __attribute__((ext_vector_type(16))) __bf16 v16bf;
typedef __attribute__((ext_vector_type(8)))  float  v8f;

union Frag {
    v16bf v;
    uint4 q[2];
};

__device__ __forceinline__ v8f wmma_bf16(const Frag& a, const Frag& b, v8f c) {
    return __builtin_amdgcn_wmma_f32_16x16x32_bf16(
        /*neg_a=*/false, a.v, /*neg_b=*/false, b.v,
        /*c_mod=*/(short)0, c, /*reuse_a=*/false, /*reuse_b=*/false);
}

// ---------------- Tensor Data Mover (guarded; fallback = direct loads) ----
#if defined(__has_builtin)
#  if __has_builtin(__builtin_amdgcn_tensor_load_to_lds)
#    define HAVE_TDM 1
#  endif
#endif
#ifndef HAVE_TDM
#  define HAVE_TDM 0
#endif
#if __has_include(<hip/amd_detail/amd_gfx1250_TDM.h>)
#  define TDM_6ARG 1
#else
#  define TDM_6ARG 0
#endif

#if HAVE_TDM
typedef __attribute__((ext_vector_type(4))) unsigned int u32x4;
typedef __attribute__((ext_vector_type(8))) int          i32x8;
typedef __attribute__((ext_vector_type(4))) int          i32x4;

// 2-D tile DMA: global (row-major, elem = 2B) -> LDS.  D# per ISA 8.3/8.4.
__device__ __forceinline__ void tdm_load_2d(const void* gptr, void* lptr,
                                            unsigned tile_d0, unsigned tile_d1,
                                            unsigned long long tens_d0,
                                            unsigned long long tens_d1,
                                            unsigned long long stride0) {
    unsigned long long ga = (unsigned long long)(uintptr_t)gptr;
    unsigned           la = (unsigned)(uintptr_t)lptr;
    u32x4 g0;
    g0.x = 1u;                                           // count=1, user desc
    g0.y = la;                                           // lds_addr
    g0.z = (unsigned)(ga & 0xffffffffu);                 // global_addr[31:0]
    g0.w = (unsigned)((ga >> 32) & 0x01ffffffu)          // global_addr[56:32]
         | (2u << 30);                                   // type = 2 ("image")
    i32x8 g1;
    g1[0] = (int)(1u << 16);                             // data_size=1 (2B)
    g1[1] = (int)((tens_d0 & 0xffffu) << 16);            // tensor_dim0[15:0]
    g1[2] = (int)(((tens_d0 >> 16) & 0xffffu)            // tensor_dim0[31:16]
                | ((tens_d1 & 0xffffu) << 16));          // tensor_dim1[15:0]
    g1[3] = (int)(((tens_d1 >> 16) & 0xffffu)            // tensor_dim1[31:16]
                | ((tile_d0 & 0xffffu) << 16));          // tile_dim0
    g1[4] = (int)(tile_d1 & 0xffffu);                    // tile_dim1 (dim2=0)
    g1[5] = (int)(stride0 & 0xffffffffu);                // dim0_stride[31:0]
    g1[6] = (int)((stride0 >> 32) & 0xffffu);            // dim0_stride[47:32]
    g1[7] = 0;
    i32x4 z4 = {0, 0, 0, 0};
#if TDM_6ARG
    i32x8 z8 = {0, 0, 0, 0, 0, 0, 0, 0};
    __builtin_amdgcn_tensor_load_to_lds(g0, g1, z4, z4, z8, 0);
#else
    __builtin_amdgcn_tensor_load_to_lds(g0, g1, z4, z4, 0);
#endif
}
#endif  // HAVE_TDM

// =====================================================================
// k1: wbf = bf16(w); w2[p] = sum_c w[p,c]^2.  512 blocks, 1 wave.
// =====================================================================
__global__ __launch_bounds__(32) void k1_prep(const float* __restrict__ w,
                                              __bf16* __restrict__ wbf,
                                              float* __restrict__ w2) {
    const int p    = blockIdx.x;
    const int lane = threadIdx.x;
    const float* row = w + (size_t)p * NC;
    float ss = 0.f;
    for (int i = lane; i < NC; i += 32) {
        float f = row[i];
        ss += f * f;
        wbf[(size_t)p * NC + i] = (__bf16)f;
    }
    for (int o = 1; o < 32; o <<= 1) ss += __shfl_xor(ss, o, 32);
    if (lane == 0) w2[p] = ss;
}

// =====================================================================
// k2: fused  S = x @ w^T -> softmax(2S - ||w||^2) -> threshold.
//     Writes assignT [P][M] bf16, xbfT [C][M] bf16, partial assign_sum.
//     One block = 16 rows, all 512 codes; 4 waves x 8 col-tiles.
// =====================================================================
__global__ __launch_bounds__(128) void k2_gemm_softmax(
        const float* __restrict__ fm, const __bf16* __restrict__ wbf,
        const float* __restrict__ w2, __bf16* __restrict__ assignT,
        __bf16* __restrict__ xbfT, float* __restrict__ assign_sum) {
    __shared__ __bf16 xt[16][NC + 8];   // padded: spread column writes on banks
    __shared__ float  St[16][NP];

    const int tid = threadIdx.x;
    const int m0  = blockIdx.x * 16;

    // stage x tile: x[m,c] = fm[b, c, hw]; lanes walk hw (coalesced reads)
    for (int i = tid; i < 16 * NC; i += 128) {
        int ml = i & 15;
        int c  = i >> 4;
        int m  = m0 + ml;
        int b  = m / NHW;
        int hw = m - b * NHW;
        xt[ml][c] = (__bf16)fm[((size_t)b * NC + c) * NHW + hw];
    }
    __syncthreads();

    const int wave = tid >> 5;
    const int lane = tid & 31;
    const int lm   = lane & 15;
    const int lh   = lane >> 4;

    v8f acc[8];
    const v8f zero = {0.f, 0.f, 0.f, 0.f, 0.f, 0.f, 0.f, 0.f};
    for (int t = 0; t < 8; ++t) acc[t] = zero;

    for (int ks = 0; ks < 16; ++ks) {
        const int c0 = ks * 32;
        Frag fa;
        fa.q[0] = *(const uint4*)&xt[lm][c0 + lh * 8];
        fa.q[1] = *(const uint4*)&xt[lm][c0 + lh * 8 + 16];
        for (int t = 0; t < 8; ++t) {
            const int p = wave * 128 + t * 16 + lm;
            const __bf16* wrow = wbf + (size_t)p * NC + c0 + lh * 8;
            Frag fb;
            fb.q[0] = *(const uint4*)wrow;
            fb.q[1] = *(const uint4*)(wrow + 16);
            acc[t] = wmma_bf16(fa, fb, acc[t]);
        }
    }

    for (int t = 0; t < 8; ++t) {
        const int p = wave * 128 + t * 16 + lm;
        for (int r = 0; r < 8; ++r)
            St[r + lh * 8][p] = acc[t][r];
    }
    __syncthreads();

    // softmax over p; ||x||^2 is row-constant and cancels.
    for (int rr = 0; rr < 2; ++rr) {
        const int ml = (tid >> 4) + rr * 8;
        const int pl = tid & 15;
        float vals[32];
        float mx = -3.4e38f;
        for (int j = 0; j < 32; ++j) {
            const int p = pl + j * 16;
            float v = 2.0f * St[ml][p] - w2[p];
            vals[j] = v;
            mx = fmaxf(mx, v);
        }
        for (int o = 1; o < 16; o <<= 1) mx = fmaxf(mx, __shfl_xor(mx, o, 32));
        float s = 0.f;
        for (int j = 0; j < 32; ++j) { float e = __expf(vals[j] - mx); vals[j] = e; s += e; }
        for (int o = 1; o < 16; o <<= 1) s += __shfl_xor(s, o, 32);
        const float inv = 1.0f / s;
        for (int j = 0; j < 32; ++j) {
            const int p = pl + j * 16;
            float a = vals[j] * inv;
            if (a < (1.0f / (float)NP)) a = 0.f;
            St[ml][p] = a;
        }
    }
    __syncthreads();

    // per column c(=p): write assignT + xbfT (32B contiguous) + assign_sum
    const int b0 = m0 / NHW;
    const int b1 = (m0 + 15) / NHW;
    for (int pp = 0; pp < 4; ++pp) {
        const int c = tid + pp * 128;
        union { __bf16 h[16]; uint4 q[2]; } ua, ux;
        float sums[2] = {0.f, 0.f};
        for (int ml = 0; ml < 16; ++ml) {
            float a = St[ml][c];
            ua.h[ml] = (__bf16)a;
            ux.h[ml] = xt[ml][c];
            int b = (m0 + ml) / NHW;
            sums[b - b0] += a;
        }
        uint4* da = (uint4*)(assignT + (size_t)c * NM + m0);
        da[0] = ua.q[0]; da[1] = ua.q[1];
        uint4* dx = (uint4*)(xbfT + (size_t)c * NM + m0);
        dx[0] = ux.q[0]; dx[1] = ux.q[1];
        atomicAdd(assign_sum + b0 * NP + c, sums[0] * (1.0f / (float)NHW));
        if (b1 != b0)
            atomicAdd(assign_sum + b1 * NP + c, sums[1] * (1.0f / (float)NHW));
    }
}

// =====================================================================
// k2b: assign_mean[p] = 196 * sum_b assign_sum[b,p];  new_w_ep_num out.
// =====================================================================
__global__ __launch_bounds__(512) void k2b_finalize(
        const float* __restrict__ assign_sum, const float* __restrict__ w_ep_num,
        float* __restrict__ assign_mean, float* __restrict__ out_wnum) {
    const int p = threadIdx.x;
    float s = 0.f;
    for (int b = 0; b < NB; ++b) s += assign_sum[b * NP + p];
    const float am = s * (float)NHW;
    assign_mean[p] = am;
    out_wnum[p] = w_ep_num[p] + (am > 1.0f ? am : 0.0f);
}

// =====================================================================
// k3: update_w = assign^T @ x, K = 25088; fused w_ep_sum epilogue.
//     64 blocks of 64x64 output; 4 waves x 4 col-tiles.
//     A operand: TDM double-buffered into per-wave LDS (fallback: global).
//     B operand: contiguous 16B loads from xbfT.
// =====================================================================
__global__ __launch_bounds__(128) void k3_updatew(
        const __bf16* __restrict__ xbfT, const __bf16* __restrict__ assignT,
        const float* __restrict__ assign_mean, const float* __restrict__ w_ep_sum,
        float* __restrict__ out_wsum) {
    const int bx   = blockIdx.x;
    const int nb   = (bx >> 3) * 64;
    const int cb   = (bx & 7) * 64;
    const int tid  = threadIdx.x;
    const int wave = tid >> 5;
    const int lane = tid & 31;
    const int lm   = lane & 15;
    const int lh   = lane >> 4;
    const int n    = nb + wave * 16 + lm;    // A row this lane feeds

    v8f acc[4];
    const v8f zero = {0.f, 0.f, 0.f, 0.f, 0.f, 0.f, 0.f, 0.f};
    for (int t = 0; t < 4; ++t) acc[t] = zero;

#if HAVE_TDM
    // per-wave double-buffered A tiles: 16 rows x 32 K, bf16 (8 KB total)
    __shared__ __bf16 atile[4][2][16][32];
    const __bf16* arow0 = assignT + (size_t)(nb + wave * 16) * NM;
    tdm_load_2d(arow0, &atile[wave][0][0][0], /*tile*/32, 16,
                /*tensor*/NM, NP, /*stride0*/NM);
#else
    const __bf16* arow_base = assignT + (size_t)n * NM + lh * 8;
#endif

    for (int ks = 0; ks < NM / 32; ++ks) {
        const int k0 = ks * 32;

#if HAVE_TDM
        // WAR guard: prior ds reads of the target buffer must retire first
        asm volatile("s_wait_dscnt 0x0" ::: "memory");
        // prefetch next K tile into the other buffer (OOB rows read as 0)
        tdm_load_2d(arow0 + (k0 + 32), &atile[wave][(ks + 1) & 1][0][0],
                    32, 16, NM, NP, NM);
        __builtin_amdgcn_s_wait_tensorcnt(1);   // current buffer ready
        Frag fa;
        fa.q[0] = *(const uint4*)&atile[wave][ks & 1][lm][lh * 8];
        fa.q[1] = *(const uint4*)&atile[wave][ks & 1][lm][lh * 8 + 16];
#else
        Frag fa;
        const __bf16* arow = arow_base + k0;
        fa.q[0] = *(const uint4*)arow;
        fa.q[1] = *(const uint4*)(arow + 16);
        __builtin_prefetch(arow + 128, 0, 1);
#endif

        for (int t = 0; t < 4; ++t) {
            const int c = cb + t * 16 + lm;
            const __bf16* brow = xbfT + (size_t)c * NM + k0 + lh * 8;
            Frag fb;
            fb.q[0] = *(const uint4*)brow;
            fb.q[1] = *(const uint4*)(brow + 16);
            if (t == 0) __builtin_prefetch(brow + 256, 0, 1);
            acc[t] = wmma_bf16(fa, fb, acc[t]);
        }
    }

    // epilogue: new_w_ep_sum = w_ep_sum + (assign_mean[n] > 1 ? update_w : 0)
    for (int t = 0; t < 4; ++t) {
        for (int r = 0; r < 8; ++r) {
            const int nrow = nb + wave * 16 + r + lh * 8;
            const int ccol = cb + t * 16 + lm;
            const float upd = (assign_mean[nrow] > 1.0f) ? acc[t][r] : 0.0f;
            out_wsum[nrow * NC + ccol] = w_ep_sum[nrow * NC + ccol] + upd;
        }
    }
}

// =====================================================================
// k4: cvec = relu((assign_sum @ w) / P); out = concat(embedding, cvec).
// =====================================================================
__global__ __launch_bounds__(32) void k4_cvec(
        const float* __restrict__ assign_sum, const __bf16* __restrict__ wbf,
        const float* __restrict__ embedding, const int* __restrict__ epoch,
        float* __restrict__ out_emb) {
    const int bx  = blockIdx.x;
    const int mb  = (bx >> 5) * 16;
    const int cbt = (bx & 31) * 16;
    const int lane = threadIdx.x;
    const int lm = lane & 15;
    const int lh = lane >> 4;

    v8f acc = {0.f, 0.f, 0.f, 0.f, 0.f, 0.f, 0.f, 0.f};

    for (int ks = 0; ks < 16; ++ks) {
        const int p0 = ks * 32;
        Frag fa, fb;
        {   // A: assign_sum rows, f32 -> bf16
            union { __bf16 h[16]; uint4 q[2]; } u;
            const float* arow = assign_sum + (mb + lm) * NP + p0 + lh * 8;
            for (int i = 0; i < 8; ++i) u.h[i]     = (__bf16)arow[i];
            for (int i = 0; i < 8; ++i) u.h[8 + i] = (__bf16)arow[16 + i];
            fa.q[0] = u.q[0]; fa.q[1] = u.q[1];
        }
        {   // B: w[p, c], column per lane
            union { __bf16 h[16]; uint4 q[2]; } u;
            const int c = cbt + lm;
            for (int i = 0; i < 8; ++i)
                u.h[i]     = wbf[(size_t)(p0 + lh * 8 + i) * NC + c];
            for (int i = 0; i < 8; ++i)
                u.h[8 + i] = wbf[(size_t)(p0 + 16 + lh * 8 + i) * NC + c];
            fb.q[0] = u.q[0]; fb.q[1] = u.q[1];
        }
        acc = wmma_bf16(fa, fb, acc);
    }

    const bool ep = (*epoch) > 2;
    for (int r = 0; r < 8; ++r) {
        const int m = mb + r + lh * 8;
        const int c = cbt + lm;
        const float cv = ep ? fmaxf(acc[r] * (1.0f / (float)NP), 0.0f) : 0.0f;
        out_emb[m * 1024 + 512 + c] = cv;
        out_emb[m * 1024 + c]       = embedding[m * NP + c];
    }
}

// =====================================================================
extern "C" void kernel_launch(void* const* d_in, const int* in_sizes, int n_in,
                              void* d_out, int out_size, void* d_ws, size_t ws_size,
                              hipStream_t stream) {
    (void)in_sizes; (void)n_in; (void)out_size; (void)ws_size;
    const float* fm        = (const float*)d_in[0];
    const float* embedding = (const float*)d_in[1];
    const float* w         = (const float*)d_in[2];
    const float* w_ep_sum  = (const float*)d_in[3];
    const float* w_ep_num  = (const float*)d_in[4];
    const int*   epoch     = (const int*)d_in[5];

    char*   ws    = (char*)d_ws;
    __bf16* wbf   = (__bf16*)(ws + OFF_WBF);
    float*  w2    = (float*)(ws + OFF_W2);
    float*  asum  = (float*)(ws + OFF_ASUM);
    __bf16* aT    = (__bf16*)(ws + OFF_AT);
    float*  amean = (float*)(ws + OFF_AMEAN);
    __bf16* xbfT  = (__bf16*)(ws + OFF_XBFT);
    float*  out   = (float*)d_out;

    hipMemsetAsync(asum, 0, (size_t)NB * NP * sizeof(float), stream);

    k1_prep<<<NP, 32, 0, stream>>>(w, wbf, w2);
    k2_gemm_softmax<<<NM / 16, 128, 0, stream>>>(fm, wbf, w2, aT, xbfT, asum);
    k2b_finalize<<<1, NP, 0, stream>>>(asum, w_ep_num, amean, out + OUT_WNUM);
    k3_updatew<<<64, 128, 0, stream>>>(xbfT, aT, amean, w_ep_sum, out + OUT_WSUM);
    k4_cvec<<<256, 32, 0, stream>>>(asum, wbf, embedding, epoch, out + OUT_EMB);
}
